// SecondOrderDecoder_5059471474979
// MI455X (gfx1250) — matrix operand order repaired and checked by reading, not compiled, not run on hardware
//
#include <hip/hip_runtime.h>
#include <stdint.h>

typedef _Float16 half_t;
typedef __attribute__((ext_vector_type(16))) half_t v16h;
typedef __attribute__((ext_vector_type(8)))  float  v8f;
typedef __attribute__((ext_vector_type(4)))  unsigned int v4u;
typedef __attribute__((ext_vector_type(8)))  int v8i;
typedef __attribute__((ext_vector_type(4)))  int v4i;

#define BATCH 8
#define NN    256
#define CINCH 32
#define HID   64
#define FOUT  32
#define CND   16
#define EPSF  1e-5f

// ---------------- TDM helper: 1-row 2D tile copy global->LDS ----------------
__device__ inline uint32_t lds_off(const void* p) {
  return (uint32_t)(uintptr_t)p;   // generic LDS addr: low 32 bits = LDS offset
}

__device__ inline void tdm_load(uint32_t lds_byte_addr, const void* gptr, uint32_t nhalves) {
  uint64_t ga = (uint64_t)gptr;
  v4u g0;
  g0[0] = 1u;                                   // count=1, user descriptor
  g0[1] = lds_byte_addr;                        // lds_addr
  g0[2] = (uint32_t)(ga & 0xffffffffu);         // global_addr[31:0]
  g0[3] = (uint32_t)((ga >> 32) & 0x01ffffffu) | (2u << 30); // addr[56:32], type=2
  v8i g1;
  g1[0] = (1 << 16);                            // data_size=1 (2 bytes/elt)
  g1[1] = (int)((nhalves & 0xffffu) << 16);     // tensor_dim0[15:0] in [31:16]
  g1[2] = (int)((nhalves >> 16) & 0xffffu) | (1 << 16); // tensor_dim0 hi, tensor_dim1=1
  g1[3] = (int)((nhalves & 0xffffu) << 16);     // tile_dim0 in [31:16]
  g1[4] = 1;                                    // tile_dim1=1, tile_dim2=0
  g1[5] = (int)nhalves;                         // tensor_dim0_stride lo32
  g1[6] = 0;
  g1[7] = 0;
  v4i z4 = {0, 0, 0, 0};
  v8i z8 = {0, 0, 0, 0, 0, 0, 0, 0};
  // 6-arg form (clang-23 / therock headers): groups 2,3 disabled, extra group zeroed
  __builtin_amdgcn_tensor_load_to_lds(g0, g1, z4, z4, z8, 0);
}

// ---------------- fragment helpers ----------------
// A-fragment half h -> K = base + h, base = ks*32 + (lane>=16)*16
__device__ inline v16h ld16(const half_t* row, int base) {
  v16h r;
#pragma unroll
  for (int h = 0; h < 16; ++h) {
    int c = base + h;
    r[h] = row[c];
  }
  return r;
}
__device__ inline v16h relu_add16(v16h a, v16h b) {
  v16h r;
#pragma unroll
  for (int h = 0; h < 16; ++h) {
    half_t v = (half_t)(a[h] + b[h]);
    r[h] = v > (half_t)0 ? v : (half_t)0;
  }
  return r;
}
__device__ inline v16h act16(const half_t* row, const float* sS, const float* sT, int base) {
  v16h r;
#pragma unroll
  for (int h = 0; h < 16; ++h) {
    int c = base + h;
    float v = (float)row[c] * sS[c] + sT[c];
    r[h] = (half_t)(v > 0.f ? v : 0.f);
  }
  return r;
}

// ---------------- K0: weight prep (fp16 transposed, l1 folded weights) -------
__global__ void __launch_bounds__(256) k_prep(
    const float* l1_wid, const float* l1_wtr,
    const float* l2_wid, const float* l2_wtr,
    const float* fc_wid, const float* fc_wtr,
    float* WU, float* WV, half_t* W2ID, half_t* W2TR, half_t* W3ID, half_t* W3TR) {
  int tid = threadIdx.x;
  for (int idx = tid; idx < CINCH * HID; idx += 256) {
    int c = idx / HID, o = idx % HID;
    WU[c * HID + o] = l1_wid[c * HID + o] + l1_wtr[(CINCH + c) * HID + o];
    WV[c * HID + o] = l1_wid[(CINCH + c) * HID + o] + l1_wtr[c * HID + o];
  }
  for (int idx = tid; idx < HID * HID; idx += 256) {     // [o][c] transposed fp16
    int o = idx >> 6, c = idx & 63;
    W2ID[idx] = (half_t)l2_wid[c * HID + o];
    W2TR[idx] = (half_t)l2_wtr[c * HID + o];
  }
  for (int idx = tid; idx < FOUT * HID; idx += 256) {
    int o = idx >> 6, c = idx & 63;
    W3ID[idx] = (half_t)fc_wid[c * FOUT + o];
    W3TR[idx] = (half_t)fc_wtr[c * FOUT + o];
  }
}

// ---------------- K1: layer1 collapses to U[b,i,o] + V[b,j,o] ----------------
__global__ void __launch_bounds__(256) k1_uv(
    const float* X, const float* WU, const float* WV,
    const float* WROW1, const float* WCOL1, const float* WG1,
    float* U1, float* V1, float* PST) {
  __shared__ float sM[CINCH];
  __shared__ float sG[HID];
  __shared__ float sRed[256];
  __shared__ float sX[256 * CINCH];
  int b = blockIdx.x, tid = threadIdx.x;
  if (tid < CINCH) sM[tid] = 0.f;
  sRed[tid] = 0.f;
  __syncthreads();
  const float* xr = X + (b * NN + tid) * CINCH;
#pragma unroll
  for (int c = 0; c < CINCH; ++c) {
    float v = xr[c];
    sX[tid * CINCH + c] = v;
    unsafeAtomicAdd(&sM[c], v);
  }
  __syncthreads();
  if (tid < CINCH) sM[tid] *= (1.f / NN);
  __syncthreads();
  if (tid < HID) {
    int o = tid;
    float g = 0.f;
    for (int c = 0; c < CINCH; ++c)
      g += sM[c] * (WG1[c * HID + o] + WG1[(CINCH + c) * HID + o] +
                    WG1[(2 * CINCH + c) * HID + o] + WG1[(3 * CINCH + c) * HID + o]);
    sG[o] = g;
  }
  __syncthreads();
  int i = tid;
  for (int o = 0; o < HID; ++o) {
    float u = 0.f, v = 0.f;
    for (int c = 0; c < CINCH; ++c) {
      float xc = sX[i * CINCH + c];
      u += xc * WU[c * HID + o];
      v += xc * WV[c * HID + o];
    }
    float fr = 0.f, fc = 0.f;
    for (int q = 0; q < 192; ++q) {
      float f;
      if (q < 32)       f = sM[q];
      else if (q < 64)  f = sX[i * CINCH + (q - 32)];
      else if (q < 96)  f = sX[i * CINCH + (q - 64)];
      else if (q < 128) f = sM[q - 96];
      else if (q < 160) f = sX[i * CINCH + (q - 128)];
      else              f = sM[q - 160];
      fr += f * WROW1[q * HID + o];
      fc += f * WCOL1[q * HID + o];
    }
    float uf = u + fr;
    float vf = v + fc + sG[o];
    U1[(b * NN + i) * HID + o] = uf;
    V1[(b * NN + i) * HID + o] = vf;
    unsafeAtomicAdd(&sRed[o], uf);
    unsafeAtomicAdd(&sRed[64 + o], uf * uf);
    unsafeAtomicAdd(&sRed[128 + o], vf);
    unsafeAtomicAdd(&sRed[192 + o], vf * vf);
  }
  __syncthreads();
  PST[b * 256 + tid] = sRed[tid] * (1.f / NN);
}

// layer1 BN stats are analytic: E[(U_i+V_j)] etc. over (b,i,j)
__global__ void __launch_bounds__(64) k1_bn(
    const float* PST, const float* g1, const float* b1, float* S1, float* T1) {
  int o = threadIdx.x;
  float mu = 0.f, e2 = 0.f;
  for (int b = 0; b < BATCH; ++b) {
    float mU = PST[b * 256 + o], mU2 = PST[b * 256 + 64 + o];
    float mV = PST[b * 256 + 128 + o], mV2 = PST[b * 256 + 192 + o];
    mu += mU + mV;
    e2 += mU2 + 2.f * mU * mV + mV2;
  }
  mu *= (1.f / BATCH);
  e2 *= (1.f / BATCH);
  float var = e2 - mu * mu;
  float s = g1[o] * rsqrtf(var + EPSF);
  S1[o] = s;
  T1[o] = b1[o] - s * mu;
}

// PUh = s1*U, PVh = s1*V + t1  (x2_l2[b,p,q,c] = relu(PU[p]+PV[q]))
__global__ void __launch_bounds__(256) k2_pre(
    const float* U1, const float* V1, const float* S1, const float* T1,
    half_t* PUH, half_t* PVH) {
  int b = blockIdx.x, tid = threadIdx.x;
  for (int idx = tid; idx < NN * HID; idx += 256) {
    int c = idx & 63;
    PUH[b * NN * HID + idx] = (half_t)(S1[c] * U1[b * NN * HID + idx]);
    PVH[b * NN * HID + idx] = (half_t)(S1[c] * V1[b * NN * HID + idx] + T1[c]);
  }
}

// row/col contractions of layer2 input (sum of relu over one axis)
__global__ void __launch_bounds__(256) k2_contract(
    const float* U1, const float* V1, const float* S1, const float* T1,
    float* R2, float* C2) {
  int b = blockIdx.y, mode = blockIdx.z, tid = threadIdx.x;
  int kl = tid >> 6, c = tid & 63;
  int k = blockIdx.x * 4 + kl;
  float s = S1[c], t = T1[c];
  float acc = 0.f;
  if (mode == 0) {
    float fixv = s * V1[(b * NN + k) * HID + c] + t;
    for (int i2 = 0; i2 < NN; ++i2) {
      float v = s * U1[(b * NN + i2) * HID + c] + fixv;
      acc += v > 0.f ? v : 0.f;
    }
    R2[(b * NN + k) * HID + c] = acc;
  } else {
    float fixv = s * U1[(b * NN + k) * HID + c];
    for (int j2 = 0; j2 < NN; ++j2) {
      float v = fixv + s * V1[(b * NN + j2) * HID + c] + t;
      acc += v > 0.f ? v : 0.f;
    }
    C2[(b * NN + k) * HID + c] = acc;
  }
}

__global__ void __launch_bounds__(64) k2_gvec(
    const float* R2, const float* U1, const float* V1, const float* S1,
    const float* T1, const float* WG2, float* G2) {
  __shared__ float sA[HID], sB[HID];
  int b = blockIdx.x, c = threadIdx.x;
  float s = S1[c], t = T1[c];
  float a = 0.f, d = 0.f;
  for (int j = 0; j < NN; ++j) a += R2[(b * NN + j) * HID + c];
  for (int k = 0; k < NN; ++k) {
    float v = s * (U1[(b * NN + k) * HID + c] + V1[(b * NN + k) * HID + c]) + t;
    d += v > 0.f ? v : 0.f;
  }
  sA[c] = a * (1.f / (NN * (float)NN));
  sB[c] = d * (1.f / NN);
  __syncthreads();
  float g = 0.f;
  for (int q = 0; q < HID; ++q)
    g += sA[q] * WG2[q * HID + c] + sB[q] * WG2[(HID + q) * HID + c];
  G2[b * HID + c] = g;
}

__global__ void __launch_bounds__(256) k2_rowcol(
    const float* R2, const float* C2, const float* U1, const float* V1,
    const float* S1, const float* T1, const float* G2,
    const float* WROW2, const float* WCOL2, float* RA2, float* CA2) {
  __shared__ float sF[4 * 192];
  int b = blockIdx.y, tid = threadIdx.x;
  int k0 = blockIdx.x * 4;
  for (int idx = tid; idx < 768; idx += 256) {
    int kl = idx / 192, q = idx % 192;
    int k = k0 + kl;
    float v;
    if (q < 64)       v = R2[(b * NN + k) * HID + q] * (1.f / NN);
    else if (q < 128) v = C2[(b * NN + k) * HID + (q - 64)] * (1.f / NN);
    else {
      int c = q - 128;
      float x = S1[c] * (U1[(b * NN + k) * HID + c] + V1[(b * NN + k) * HID + c]) + T1[c];
      v = x > 0.f ? x : 0.f;
    }
    sF[idx] = v;
  }
  __syncthreads();
  int kl = tid >> 6, o = tid & 63;
  int k = k0 + kl;
  float ra = G2[b * HID + o], ca = 0.f;
  for (int q = 0; q < 192; ++q) {
    float f = sF[kl * 192 + q];
    ra += f * WROW2[q * HID + o];
    ca += f * WCOL2[q * HID + o];
  }
  RA2[(b * NN + k) * HID + o] = ra;
  CA2[(b * NN + k) * HID + o] = ca;
}

// ---------------- K2 main GEMM: WMMA, A-tiles generated from LDS PU/PV ------
__global__ void __launch_bounds__(256) k2_gemm(
    const float* RA2, const float* CA2, const half_t* PUH, const half_t* PVH,
    const half_t* W2ID, const half_t* W2TR, half_t* Y2H, float* ST2) {
  __shared__ half_t sPU[NN * HID];
  __shared__ half_t sPV[NN * HID];
  __shared__ float sStat[2 * HID];
  int b = blockIdx.y, i = blockIdx.x, tid = threadIdx.x;
  if (tid < 2 * HID) sStat[tid] = 0.f;
  if (tid < 32) {   // wave 0 drives the Tensor Data Mover
    tdm_load(lds_off(sPU), PUH + (size_t)b * NN * HID, NN * HID);
    tdm_load(lds_off(sPV), PVH + (size_t)b * NN * HID, NN * HID);
    __builtin_amdgcn_s_wait_tensorcnt(0);
  }
  __syncthreads();
  int lane = tid & 31, w = tid >> 5;
  int col = lane & 15, hs = lane >> 4;
  v16h pui[2], pvi[2];
#pragma unroll
  for (int ks = 0; ks < 2; ++ks) {
    int base = ks * 32 + hs * 16;
    pui[ks] = ld16(sPU + i * HID, base);
    pvi[ks] = ld16(sPV + i * HID, base);
  }
#pragma unroll
  for (int ti = 0; ti < 2; ++ti) {
    int t = w + ti * 8;
    int j0 = t * 16;
    int jr = j0 + col;
    v16h pvj[2], puj[2];
#pragma unroll
    for (int ks = 0; ks < 2; ++ks) {
      int base = ks * 32 + hs * 16;
      pvj[ks] = ld16(sPV + jr * HID, base);
      puj[ks] = ld16(sPU + jr * HID, base);
    }
    v16h a1k0 = relu_add16(pui[0], pvj[0]);   // x2[b,i,j,:]
    v16h a1k1 = relu_add16(pui[1], pvj[1]);
    v16h a2k0 = relu_add16(puj[0], pvi[0]);   // x2[b,j,i,:]
    v16h a2k1 = relu_add16(puj[1], pvi[1]);
#pragma unroll
    for (int ct = 0; ct < 4; ++ct) {
      v8f acc = {};
      const half_t* wb = W2ID + ((ct * 16 + col) * HID + hs * 16);
      const half_t* wt = W2TR + ((ct * 16 + col) * HID + hs * 16);
      v16h b0 = *(const v16h*)(wb);
      v16h b1 = *(const v16h*)(wb + 32);
      v16h t0 = *(const v16h*)(wt);
      v16h t1 = *(const v16h*)(wt + 32);
      acc = __builtin_amdgcn_wmma_f32_16x16x32_f16(false, a1k0, false, b0, (short)0, acc, false, false);
      acc = __builtin_amdgcn_wmma_f32_16x16x32_f16(false, a1k1, false, b1, (short)0, acc, false, false);
      acc = __builtin_amdgcn_wmma_f32_16x16x32_f16(false, a2k0, false, t0, (short)0, acc, false, false);
      acc = __builtin_amdgcn_wmma_f32_16x16x32_f16(false, a2k1, false, t1, (short)0, acc, false, false);
      int o = ct * 16 + col;
      float ra = RA2[(b * NN + i) * HID + o];
      float sum = 0.f, ssq = 0.f;
#pragma unroll
      for (int r = 0; r < 8; ++r) {
        int m = hs * 8 + r;
        int j = j0 + m;
        float val = acc[r] + ra + CA2[(b * NN + j) * HID + o];
        Y2H[(((size_t)(b * NN) + i) * NN + j) * HID + o] = (half_t)val;
        sum += val;
        ssq += val * val;
      }
      unsafeAtomicAdd(&sStat[o], sum);
      unsafeAtomicAdd(&sStat[HID + o], ssq);
    }
  }
  __syncthreads();
  if (tid < 2 * HID) unsafeAtomicAdd(&ST2[tid], sStat[tid]);
}

__global__ void __launch_bounds__(64) k2_bn(
    const float* ST2, const float* g2, const float* b2, float* S2, float* T2) {
  int o = threadIdx.x;
  float cnt = (float)BATCH * NN * NN;
  float mu = ST2[o] / cnt;
  float var = ST2[HID + o] / cnt - mu * mu;
  float s = g2[o] * rsqrtf(var + EPSF);
  S2[o] = s;
  T2[o] = b2[o] - s * mu;
}

// contractions of layer3 input x3 = relu(s2*Y2 + t2)
__global__ void __launch_bounds__(256) k3_contract(
    const half_t* Y2H, const float* S2, const float* T2, float* R3, float* C3) {
  int b = blockIdx.y, mode = blockIdx.z, tid = threadIdx.x;
  int kl = tid >> 6, c = tid & 63;
  int k = blockIdx.x * 4 + kl;
  float s = S2[c], t = T2[c];
  float acc = 0.f;
  if (mode == 0) {
    for (int i2 = 0; i2 < NN; ++i2) {
      float v = s * (float)Y2H[(((size_t)(b * NN) + i2) * NN + k) * HID + c] + t;
      acc += v > 0.f ? v : 0.f;
    }
    R3[(b * NN + k) * HID + c] = acc;
  } else {
    for (int j2 = 0; j2 < NN; ++j2) {
      float v = s * (float)Y2H[(((size_t)(b * NN) + k) * NN + j2) * HID + c] + t;
      acc += v > 0.f ? v : 0.f;
    }
    C3[(b * NN + k) * HID + c] = acc;
  }
}

__global__ void __launch_bounds__(64) k3_gvec(
    const float* R3, const half_t* Y2H, const float* S2, const float* T2,
    const float* WG3, float* G3, float* G3AB) {
  __shared__ float sA[HID], sB[HID];
  int b = blockIdx.x, c = threadIdx.x;
  float s = S2[c], t = T2[c];
  float a = 0.f, d = 0.f;
  for (int j = 0; j < NN; ++j) a += R3[(b * NN + j) * HID + c];
  for (int k = 0; k < NN; ++k) {
    float v = s * (float)Y2H[(((size_t)(b * NN) + k) * NN + k) * HID + c] + t;
    d += v > 0.f ? v : 0.f;
  }
  a *= (1.f / (NN * (float)NN));
  d *= (1.f / NN);
  sA[c] = a;
  sB[c] = d;
  G3AB[b * 128 + c] = a;
  G3AB[b * 128 + 64 + c] = d;
  __syncthreads();
  if (c < FOUT) {
    float g = 0.f;
    for (int q = 0; q < HID; ++q)
      g += sA[q] * WG3[q * FOUT + c] + sB[q] * WG3[(HID + q) * FOUT + c];
    G3[b * FOUT + c] = g;
  }
}

__global__ void __launch_bounds__(256) k3_rowcol(
    const float* R3, const float* C3, const half_t* Y2H, const float* S2,
    const float* T2, const float* G3, const float* WROW3, const float* WCOL3,
    float* RA3, float* CA3) {
  __shared__ float sF[8 * 192];
  int b = blockIdx.y, tid = threadIdx.x;
  int k0 = blockIdx.x * 8;
  for (int idx = tid; idx < 8 * 192; idx += 256) {
    int kl = idx / 192, q = idx % 192;
    int k = k0 + kl;
    float v;
    if (q < 64)       v = R3[(b * NN + k) * HID + q] * (1.f / NN);
    else if (q < 128) v = C3[(b * NN + k) * HID + (q - 64)] * (1.f / NN);
    else {
      int c = q - 128;
      float x = S2[c] * (float)Y2H[(((size_t)(b * NN) + k) * NN + k) * HID + c] + T2[c];
      v = x > 0.f ? x : 0.f;
    }
    sF[idx] = v;
  }
  __syncthreads();
  int kl = tid >> 5, o = tid & 31;
  int k = k0 + kl;
  float ra = G3[b * FOUT + o], ca = 0.f;
  for (int q = 0; q < 192; ++q) {
    float f = sF[kl * 192 + q];
    ra += f * WROW3[q * FOUT + o];
    ca += f * WCOL3[q * FOUT + o];
  }
  RA3[(b * NN + k) * FOUT + o] = ra;
  CA3[(b * NN + k) * FOUT + o] = ca;
}

// ---------------- K3 GEMM: fc layer via WMMA, Y2 activated on load ----------
__global__ void __launch_bounds__(256) k3_gemm(
    const float* RA3, const float* CA3, const half_t* Y2H, const float* S2,
    const float* T2, const half_t* W3ID, const half_t* W3TR, float* OUT, float* ST3) {
  __shared__ float sS[HID], sT[HID], sStat[2 * FOUT];
  int b = blockIdx.y, i = blockIdx.x, tid = threadIdx.x;
  if (tid < HID) { sS[tid] = S2[tid]; sT[tid] = T2[tid]; }
  if (tid < 2 * FOUT) sStat[tid] = 0.f;
  __syncthreads();
  int lane = tid & 31, w = tid >> 5;
  int col = lane & 15, hs = lane >> 4;
  const half_t* rowI = Y2H + (((size_t)(b * NN) + i) * NN) * HID;
#pragma unroll
  for (int ti = 0; ti < 2; ++ti) {
    int t = w + ti * 8;
    int j0 = t * 16;
    int jr = j0 + col;
    v16h a1[2], a2[2];
#pragma unroll
    for (int ks = 0; ks < 2; ++ks) {
      int base = ks * 32 + hs * 16;
      a1[ks] = act16(rowI + (size_t)jr * HID, sS, sT, base);                          // x2[b,i,j,:]
      a2[ks] = act16(Y2H + (((size_t)(b * NN) + jr) * NN + i) * HID, sS, sT, base);   // x2[b,j,i,:]
    }
#pragma unroll
    for (int ct = 0; ct < 2; ++ct) {
      v8f acc = {};
      const half_t* wb = W3ID + ((ct * 16 + col) * HID + hs * 16);
      const half_t* wt = W3TR + ((ct * 16 + col) * HID + hs * 16);
      v16h b0 = *(const v16h*)(wb);
      v16h b1 = *(const v16h*)(wb + 32);
      v16h t0 = *(const v16h*)(wt);
      v16h t1 = *(const v16h*)(wt + 32);
      acc = __builtin_amdgcn_wmma_f32_16x16x32_f16(false, a1[0], false, b0, (short)0, acc, false, false);
      acc = __builtin_amdgcn_wmma_f32_16x16x32_f16(false, a1[1], false, b1, (short)0, acc, false, false);
      acc = __builtin_amdgcn_wmma_f32_16x16x32_f16(false, a2[0], false, t0, (short)0, acc, false, false);
      acc = __builtin_amdgcn_wmma_f32_16x16x32_f16(false, a2[1], false, t1, (short)0, acc, false, false);
      int o = ct * 16 + col;
      float ra = RA3[(b * NN + i) * FOUT + o];
      float sum = 0.f, ssq = 0.f;
#pragma unroll
      for (int r = 0; r < 8; ++r) {
        int m = hs * 8 + r;
        int j = j0 + m;
        float val = acc[r] + ra + CA3[(b * NN + j) * FOUT + o];
        OUT[(((size_t)(b * NN) + i) * NN + j) * FOUT + o] = val;   // pre-BN, fixed by k4
        sum += val;
        ssq += val * val;
      }
      unsafeAtomicAdd(&sStat[o], sum);
      unsafeAtomicAdd(&sStat[FOUT + o], ssq);
    }
  }
  __syncthreads();
  if (tid < 2 * FOUT) unsafeAtomicAdd(&ST3[tid], sStat[tid]);
}

__global__ void __launch_bounds__(32) k3_bn(
    const float* ST3, const float* g3, const float* b3, float* AB3) {
  int o = threadIdx.x;
  float cnt = (float)BATCH * NN * NN;
  float mu = ST3[o] / cnt;
  float var = ST3[FOUT + o] / cnt - mu * mu;
  float a = g3[o] * rsqrtf(var + EPSF);
  AB3[o] = a;
  AB3[FOUT + o] = b3[o] - a * mu;
}

__global__ void __launch_bounds__(256) k4_final(float* OUT, const float* AB3) {
  size_t idx = (size_t)blockIdx.x * 256 + threadIdx.x;   // float4 granularity
  float4* p = (float4*)OUT;
  float4 v = p[idx];
  int o = (int)((idx * 4) & 31);
  v.x = AB3[o] * v.x + AB3[FOUT + o];
  v.y = AB3[o + 1] * v.y + AB3[FOUT + o + 1];
  v.z = AB3[o + 2] * v.z + AB3[FOUT + o + 2];
  v.w = AB3[o + 3] * v.w + AB3[FOUT + o + 3];
  p[idx] = v;
}

// ---------------- node head ----------------
__global__ void __launch_bounds__(256) k5_node(
    const float* R3, const float* C3, const half_t* Y2H, const float* S2,
    const float* T2, const float* G3AB, const float* UR, const float* UC,
    const float* UD, const float* UG, float* YN, float* STN) {
  __shared__ float sGn[CND];
  __shared__ float sSt[2 * CND];
  int b = blockIdx.x, tid = threadIdx.x;
  if (tid < 2 * CND) sSt[tid] = 0.f;
  if (tid < CND) {
    float g = 0.f;
    for (int q = 0; q < HID; ++q)
      g += G3AB[b * 128 + q] * UG[q * CND + tid] +
           G3AB[b * 128 + 64 + q] * UG[(HID + q) * CND + tid];
    sGn[tid] = g;
  }
  __syncthreads();
  int k = tid;
  float acc[CND];
#pragma unroll
  for (int u = 0; u < CND; ++u) acc[u] = 0.f;
  for (int c = 0; c < HID; ++c) {
    float r = R3[(b * NN + k) * HID + c] * (1.f / NN);
    float cc = C3[(b * NN + k) * HID + c] * (1.f / NN);
    float dvv = S2[c] * (float)Y2H[(((size_t)(b * NN) + k) * NN + k) * HID + c] + T2[c];
    float dv = dvv > 0.f ? dvv : 0.f;
#pragma unroll
    for (int u = 0; u < CND; ++u)
      acc[u] += r * UR[c * CND + u] + cc * UC[c * CND + u] + dv * UD[c * CND + u];
  }
#pragma unroll
  for (int u = 0; u < CND; ++u) {
    float y = acc[u] + sGn[u];
    YN[(b * NN + k) * CND + u] = y;
    unsafeAtomicAdd(&sSt[u], y);
    unsafeAtomicAdd(&sSt[CND + u], y * y);
  }
  __syncthreads();
  if (tid < 2 * CND) unsafeAtomicAdd(&STN[tid], sSt[tid]);
}

__global__ void __launch_bounds__(256) k5_final(
    const float* STN, const float* YN, const float* ncg, const float* ncb, float* OUT) {
  __shared__ float sAB[2 * CND];
  int tid = threadIdx.x;
  if (tid < CND) {
    float cnt = (float)BATCH * NN;
    float mu = STN[tid] / cnt;
    float var = STN[CND + tid] / cnt - mu * mu;
    float a = ncg[tid] * rsqrtf(var + EPSF);
    sAB[tid] = a;
    sAB[CND + tid] = ncb[tid] - a * mu;
  }
  __syncthreads();
  for (int idx = tid; idx < BATCH * NN * CND; idx += 256) {
    int u = idx & (CND - 1);
    OUT[(size_t)BATCH * NN * NN * FOUT + idx] = sAB[u] * YN[idx] + sAB[CND + u];
  }
}

// ---------------- host-side orchestration ----------------
extern "C" void kernel_launch(void* const* d_in, const int* in_sizes, int n_in,
                              void* d_out, int out_size, void* d_ws, size_t ws_size,
                              hipStream_t stream) {
  const float* x       = (const float*)d_in[0];
  const float* l1_wid  = (const float*)d_in[1];
  const float* l1_wtr  = (const float*)d_in[2];
  const float* l1_wrow = (const float*)d_in[3];
  const float* l1_wcol = (const float*)d_in[4];
  const float* l1_wg   = (const float*)d_in[5];
  const float* l1_g    = (const float*)d_in[6];
  const float* l1_b    = (const float*)d_in[7];
  const float* l2_wid  = (const float*)d_in[8];
  const float* l2_wtr  = (const float*)d_in[9];
  const float* l2_wrow = (const float*)d_in[10];
  const float* l2_wcol = (const float*)d_in[11];
  const float* l2_wg   = (const float*)d_in[12];
  const float* l2_g    = (const float*)d_in[13];
  const float* l2_b    = (const float*)d_in[14];
  const float* fc_wid  = (const float*)d_in[15];
  const float* fc_wtr  = (const float*)d_in[16];
  const float* fc_wrow = (const float*)d_in[17];
  const float* fc_wcol = (const float*)d_in[18];
  const float* fc_wg   = (const float*)d_in[19];
  const float* fc_g    = (const float*)d_in[20];
  const float* fc_b    = (const float*)d_in[21];
  const float* nc_ur   = (const float*)d_in[22];
  const float* nc_uc   = (const float*)d_in[23];
  const float* nc_ud   = (const float*)d_in[24];
  const float* nc_ug   = (const float*)d_in[25];
  const float* nc_g    = (const float*)d_in[26];
  const float* nc_b    = (const float*)d_in[27];
  float* OUT = (float*)d_out;

  // workspace carve-up (floats, then fp16; layout keeps 32B alignment for v16h)
  float* ws = (float*)d_ws;
  float* U1  = ws;                 // 131072
  float* V1  = U1 + 131072;        // 131072
  float* PST = V1 + 131072;        // 2048
  float* S1  = PST + 2048;         // 64
  float* T1  = S1 + 64;            // 64
  float* R2  = T1 + 64;            // 131072
  float* C2  = R2 + 131072;        // 131072
  float* RA2 = C2 + 131072;        // 131072
  float* CA2 = RA2 + 131072;       // 131072
  float* G2  = CA2 + 131072;       // 512
  float* ST2 = G2 + 512;           // 128
  float* S2  = ST2 + 128;          // 64
  float* T2  = S2 + 64;            // 64
  float* R3  = T2 + 64;            // 131072
  float* C3  = R3 + 131072;        // 131072
  float* RA3 = C3 + 131072;        // 65536
  float* CA3 = RA3 + 65536;        // 65536
  float* G3  = CA3 + 65536;        // 256
  float* G3AB= G3 + 256;           // 1024
  float* ST3 = G3AB + 1024;        // 64
  float* AB3 = ST3 + 64;           // 64
  float* YN  = AB3 + 64;           // 32768
  float* STN = YN + 32768;         // 32
  float* WU  = STN + 32;           // 2048
  float* WV  = WU + 2048;          // 2048
  half_t* W2ID = (half_t*)(WV + 2048);  // 4096
  half_t* W2TR = W2ID + 4096;           // 4096
  half_t* W3ID = W2TR + 4096;           // 2048
  half_t* W3TR = W3ID + 2048;           // 2048
  half_t* PUH  = W3TR + 2048;           // 131072
  half_t* PVH  = PUH + 131072;          // 131072
  half_t* Y2H  = PVH + 131072;          // 33554432 (64 MB, L2-resident)

  (void)hipMemsetAsync(ST2, 0, 2 * HID * sizeof(float), stream);
  (void)hipMemsetAsync(ST3, 0, 2 * FOUT * sizeof(float), stream);
  (void)hipMemsetAsync(STN, 0, 2 * CND * sizeof(float), stream);

  k_prep<<<1, 256, 0, stream>>>(l1_wid, l1_wtr, l2_wid, l2_wtr, fc_wid, fc_wtr,
                                WU, WV, W2ID, W2TR, W3ID, W3TR);
  k1_uv<<<BATCH, 256, 0, stream>>>(x, WU, WV, l1_wrow, l1_wcol, l1_wg, U1, V1, PST);
  k1_bn<<<1, 64, 0, stream>>>(PST, l1_g, l1_b, S1, T1);
  k2_pre<<<BATCH, 256, 0, stream>>>(U1, V1, S1, T1, PUH, PVH);
  k2_contract<<<dim3(NN / 4, BATCH, 2), 256, 0, stream>>>(U1, V1, S1, T1, R2, C2);
  k2_gvec<<<BATCH, 64, 0, stream>>>(R2, U1, V1, S1, T1, l2_wg, G2);
  k2_rowcol<<<dim3(NN / 4, BATCH), 256, 0, stream>>>(R2, C2, U1, V1, S1, T1, G2,
                                                     l2_wrow, l2_wcol, RA2, CA2);
  k2_gemm<<<dim3(NN, BATCH), 256, 0, stream>>>(RA2, CA2, PUH, PVH, W2ID, W2TR, Y2H, ST2);
  k2_bn<<<1, 64, 0, stream>>>(ST2, l2_g, l2_b, S2, T2);
  k3_contract<<<dim3(NN / 4, BATCH, 2), 256, 0, stream>>>(Y2H, S2, T2, R3, C3);
  k3_gvec<<<BATCH, 64, 0, stream>>>(R3, Y2H, S2, T2, fc_wg, G3, G3AB);
  k3_rowcol<<<dim3(NN / 8, BATCH), 256, 0, stream>>>(R3, C3, Y2H, S2, T2, G3,
                                                     fc_wrow, fc_wcol, RA3, CA3);
  k3_gemm<<<dim3(NN, BATCH), 256, 0, stream>>>(RA3, CA3, Y2H, S2, T2, W3ID, W3TR, OUT, ST3);
  k3_bn<<<1, 32, 0, stream>>>(ST3, fc_g, fc_b, AB3);
  k4_final<<<(BATCH * NN * NN * FOUT) / (4 * 256), 256, 0, stream>>>(OUT, AB3);
  k5_node<<<BATCH, 256, 0, stream>>>(R3, C3, Y2H, S2, T2, G3AB, nc_ur, nc_uc,
                                     nc_ud, nc_ug, YN, STN);
  k5_final<<<1, 256, 0, stream>>>(STN, YN, nc_g, nc_b, OUT);
}
